// PointCloudDiscriminator_64991445123091
// MI455X (gfx1250) — matrix operand order repaired and checked by reading, not compiled, not run on hardware
//
#include <hip/hip_runtime.h>
#include <cstdint>
#include <cstddef>

// ---------------------------------------------------------------------------
// PointNet++ discriminator for MI455X (gfx1250, wave32).
//   * FPS / KNN: wave32 shuffle-reduction kernels (latency-bound serial part).
//   * Every 1x1-conv / FC layer: v_wmma_f32_16x16x32_f16 GEMM whose operands
//     are pre-packed into fragment-native f16 layout, so the inner loop is
//     exactly {2x global_load_b128 (A), 2x global_load_b128 (B), v_wmma}.
//   * BN (training-mode batch stats) folded to scale/shift; activation applied
//     in the vectorized A-pack pass. K=16 neighbor max-pool fused into the
//     GEMM epilogue (one 16-row M-tile == one neighbor group).
// ---------------------------------------------------------------------------

typedef __attribute__((ext_vector_type(16))) _Float16 v16h;
typedef __attribute__((ext_vector_type(8)))  float    v8f;

#define FPS_THREADS 1024

// A-fragment element -> K mapping (ISA 16-bit A 16x32 layout)
//   kk = ks*32 + (e&7) + 8*hi + 16*(e>>3)
// B-fragment element -> K mapping (ISA 16-bit B 32x16 layout)
//   kk = ks*32 + (e&7) + 8*hi + 16*(e>>3) ,  n = tn*16 + (lane&15)

// ------------------------- furthest point sampling -------------------------
__global__ __launch_bounds__(FPS_THREADS)
void fps_kernel(const float* __restrict__ xyz, int n, int npoint,
                int* __restrict__ fidx, float* __restrict__ new_xyz) {
    const int b   = blockIdx.x;
    const int tid = threadIdx.x;
    const float* P = xyz + (size_t)b * n * 3;

    const int MAXP = 8;
    float px[MAXP], py[MAXP], pz[MAXP], dist[MAXP];
    int cnt = 0;
    for (int i = tid; i < n; i += FPS_THREADS) {
        px[cnt] = P[i * 3 + 0];
        py[cnt] = P[i * 3 + 1];
        pz[cnt] = P[i * 3 + 2];
        dist[cnt] = 1.0e10f;
        ++cnt;
    }

    __shared__ float cx, cy, cz;
    __shared__ float sv[32];
    __shared__ int   si[32];
    __shared__ int   sfar;
    if (tid == 0) { cx = P[0]; cy = P[1]; cz = P[2]; sfar = 0; }
    __syncthreads();

    for (int it = 0; it < npoint; ++it) {
        const float lcx = cx, lcy = cy, lcz = cz;
        if (tid == 0) {
            fidx[b * npoint + it] = sfar;
            new_xyz[(size_t)(b * npoint + it) * 3 + 0] = lcx;
            new_xyz[(size_t)(b * npoint + it) * 3 + 1] = lcy;
            new_xyz[(size_t)(b * npoint + it) * 3 + 2] = lcz;
        }
        float bv = -1.0f;
        int   bidx = 0x7fffffff;
        int i = tid;
        for (int j = 0; j < cnt; ++j, i += FPS_THREADS) {
            const float dx = px[j] - lcx, dy = py[j] - lcy, dz = pz[j] - lcz;
            const float dd = dx * dx + dy * dy + dz * dz;
            const float nd = fminf(dist[j], dd);
            dist[j] = nd;
            if (nd > bv || (nd == bv && i < bidx)) { bv = nd; bidx = i; }
        }
        #pragma unroll
        for (int off = 16; off >= 1; off >>= 1) {
            const float ov = __shfl_down(bv, (unsigned)off, 32);
            const int   oi = __shfl_down(bidx, (unsigned)off, 32);
            if (ov > bv || (ov == bv && oi < bidx)) { bv = ov; bidx = oi; }
        }
        const int wv = tid >> 5;
        if ((tid & 31) == 0) { sv[wv] = bv; si[wv] = bidx; }
        __syncthreads();
        if (tid < 32) {
            const int nw = FPS_THREADS / 32;
            float v  = (tid < nw) ? sv[tid] : -2.0f;
            int   ix = (tid < nw) ? si[tid] : 0x7fffffff;
            #pragma unroll
            for (int off = 16; off >= 1; off >>= 1) {
                const float ov = __shfl_down(v, (unsigned)off, 32);
                const int   oi = __shfl_down(ix, (unsigned)off, 32);
                if (ov > v || (ov == v && oi < ix)) { v = ov; ix = oi; }
            }
            if (tid == 0) {
                sfar = ix;
                cx = P[ix * 3 + 0]; cy = P[ix * 3 + 1]; cz = P[ix * 3 + 2];
            }
        }
        __syncthreads();
    }
}

// --------------------------------- KNN -------------------------------------
__global__ __launch_bounds__(256)
void knn_kernel(const float* __restrict__ query, const float* __restrict__ ref,
                int S, int n, int nquery, int* __restrict__ out_idx) {
    const int wid  = blockIdx.x * (blockDim.x >> 5) + (threadIdx.x >> 5);
    const int lane = threadIdx.x & 31;
    if (wid >= nquery) return;
    const int b = wid / S;

    const float* q = query + (size_t)wid * 3;
    const float qx = q[0], qy = q[1], qz = q[2];
    const float* R = ref + (size_t)b * n * 3;

    float bd[16];
    int   bi[16];
    #pragma unroll
    for (int j = 0; j < 16; ++j) { bd[j] = 3.0e38f; bi[j] = -1; }

    for (int i = lane; i < n; i += 32) {
        const float dx = R[i * 3 + 0] - qx;
        const float dy = R[i * 3 + 1] - qy;
        const float dz = R[i * 3 + 2] - qz;
        float cd = dx * dx + dy * dy + dz * dz;
        if (cd < bd[15]) {
            int ci = i;
            #pragma unroll
            for (int j = 0; j < 16; ++j) {
                if (cd < bd[j]) {
                    const float tf = bd[j]; bd[j] = cd; cd = tf;
                    const int   ti = bi[j]; bi[j] = ci; ci = ti;
                }
            }
        }
    }
    for (int r = 0; r < 16; ++r) {
        float mv = bd[0];
        int   mi = bi[0];
        #pragma unroll
        for (int off = 16; off >= 1; off >>= 1) {
            const float ov = __shfl_down(mv, (unsigned)off, 32);
            const int   oi = __shfl_down(mi, (unsigned)off, 32);
            if (ov < mv || (ov == mv && (unsigned)oi < (unsigned)mi)) { mv = ov; mi = oi; }
        }
        mv = __shfl(mv, 0, 32);
        mi = __shfl(mi, 0, 32);
        if (lane == 0) out_idx[(size_t)wid * 16 + r] = mi;
        if (bd[0] == mv && bi[0] == mi) {
            #pragma unroll
            for (int j = 0; j < 15; ++j) { bd[j] = bd[j + 1]; bi[j] = bi[j + 1]; }
            bd[15] = 3.0e38f; bi[15] = -1;
        }
    }
}

// -------------------- packed grouped-feature builders ----------------------
// SA1: C=6 (Cpad=32, nkb=1). Row t=(b,s,kk). Only lane (t&15), e=0..5 nonzero.
__global__ void build_sa1_packed(const float* __restrict__ xyz,
                                 const float* __restrict__ new_xyz,
                                 const int* __restrict__ knn,
                                 _Float16* __restrict__ Apk, int nq, int N) {
    const int t = blockIdx.x * blockDim.x + threadIdx.x;
    if (t >= nq * 16) return;
    const int bs = t >> 4;
    const int b  = bs >> 9;  // / 512
    const int idx = knn[t];
    const float* p = xyz + ((size_t)b * N + idx) * 3;
    const float* c = new_xyz + (size_t)bs * 3;
    v16h s = {};
    s[0] = (_Float16)(p[0] - c[0]);
    s[1] = (_Float16)(p[1] - c[1]);
    s[2] = (_Float16)(p[2] - c[2]);
    s[3] = (_Float16)p[0];
    s[4] = (_Float16)p[1];
    s[5] = (_Float16)p[2];
    const int tm = t >> 4, lo = t & 15;
    v16h z = {};
    ((v16h*)Apk)[(size_t)tm * 32 + lo]      = s;   // hi=0 slot
    ((v16h*)Apk)[(size_t)tm * 32 + lo + 16] = z;   // hi=1 slot (kk>=8 -> pad)
}

// SA2: C=131 (Cpad=160, nkb=5). One thread per (row, ks), writes both hi slots.
__global__ void build_sa2_packed(const float* __restrict__ l1_xyz,
                                 const float* __restrict__ l1_p,
                                 const float* __restrict__ l2_xyz,
                                 const int* __restrict__ knn,
                                 _Float16* __restrict__ Apk,
                                 int nrows, int nkb, int S1, int S2, int C) {
    const int t = blockIdx.x * blockDim.x + threadIdx.x;
    if (t >= nrows * nkb) return;
    const int ks  = t % nkb;
    const int row = t / nkb;
    const int bs = row >> 4;
    const int b  = bs / S2;
    const int idx = knn[row];
    const float* p = l1_xyz + ((size_t)b * S1 + idx) * 3;
    const float* c = l2_xyz + (size_t)bs * 3;
    const float r0 = p[0] - c[0], r1 = p[1] - c[1], r2 = p[2] - c[2];
    const float* f = l1_p + ((size_t)b * S1 + idx) * C;  // C channels after 3 coords
    const int tm = row >> 4, lo = row & 15;
    const int NF = C + 3;
    #pragma unroll
    for (int hi = 0; hi < 2; ++hi) {
        v16h s;
        #pragma unroll
        for (int e = 0; e < 16; ++e) {
            const int kk = ks * 32 + (e & 7) + 8 * hi + 16 * (e >> 3);
            int j = kk - 3;
            j = j < 0 ? 0 : (j >= C ? C - 1 : j);
            const float fv = f[j];                       // always in-bounds load
            const float rv = (kk == 0) ? r0 : ((kk == 1) ? r1 : r2);
            const float v  = (kk < 3) ? rv : ((kk < NF) ? fv : 0.0f);
            s[e] = (_Float16)v;
        }
        ((v16h*)Apk)[((size_t)tm * nkb + ks) * 32 + lo + 16 * hi] = s;
    }
}

// SA3: C=259 (Cpad=288, nkb=9). Row = b*128 + j, feats = concat(l2_xyz, l2_p)
__global__ void build_sa3_packed(const float* __restrict__ l2_xyz,
                                 const float* __restrict__ l2_p,
                                 _Float16* __restrict__ Apk,
                                 int nrows, int nkb, int C) {
    const int t = blockIdx.x * blockDim.x + threadIdx.x;
    if (t >= nrows * nkb) return;
    const int ks  = t % nkb;
    const int row = t / nkb;
    const float r0 = l2_xyz[row * 3 + 0];
    const float r1 = l2_xyz[row * 3 + 1];
    const float r2 = l2_xyz[row * 3 + 2];
    const float* f = l2_p + (size_t)row * C;
    const int tm = row >> 4, lo = row & 15;
    const int NF = C + 3;
    #pragma unroll
    for (int hi = 0; hi < 2; ++hi) {
        v16h s;
        #pragma unroll
        for (int e = 0; e < 16; ++e) {
            const int kk = ks * 32 + (e & 7) + 8 * hi + 16 * (e >> 3);
            int j = kk - 3;
            j = j < 0 ? 0 : (j >= C ? C - 1 : j);
            const float fv = f[j];
            const float rv = (kk == 0) ? r0 : ((kk == 1) ? r1 : r2);
            const float v  = (kk < 3) ? rv : ((kk < NF) ? fv : 0.0f);
            s[e] = (_Float16)v;
        }
        ((v16h*)Apk)[((size_t)tm * nkb + ks) * 32 + lo + 16 * hi] = s;
    }
}

// -------------------- BN stats -> fused scale / shift ----------------------
__global__ __launch_bounds__(256)
void stats_kernel(const float* __restrict__ H, int M, int C,
                  const float* __restrict__ gamma, const float* __restrict__ beta,
                  float* __restrict__ scale, float* __restrict__ shift) {
    const int c = blockIdx.x;
    const int tid = threadIdx.x;
    float s = 0.0f, ss = 0.0f;
    for (int m = tid; m < M; m += 256) {
        const float v = H[(size_t)m * C + c];
        s += v; ss += v * v;
    }
    __shared__ float s1[256], s2[256];
    s1[tid] = s; s2[tid] = ss;
    __syncthreads();
    for (int off = 128; off > 0; off >>= 1) {
        if (tid < off) { s1[tid] += s1[tid + off]; s2[tid] += s2[tid + off]; }
        __syncthreads();
    }
    if (tid == 0) {
        const float mu  = s1[0] / (float)M;
        const float var = s2[0] / (float)M - mu * mu;   // biased, like jnp.var
        const float sc  = rsqrtf(var + 1e-5f) * gamma[c];
        scale[c] = sc;
        shift[c] = beta[c] - mu * sc;
    }
}

// ------------------ activation + A-fragment pack (vectorized) --------------
// X: M x Cpad f32, fully initialized, Cpad % 32 == 0. One thread per
// (tile_m, ks, lane); t itself equals the packed v16h index.
enum { ACT_NONE = 0, ACT_RELU = 1, ACT_LEAKY = 2 };

template <int ACT>
__global__ __launch_bounds__(256)
void apack_kernel(const float* __restrict__ X, int nkb, int Cpad,
                  const float* __restrict__ scale, const float* __restrict__ shift,
                  _Float16* __restrict__ Apk, int total) {
    const int t = blockIdx.x * blockDim.x + threadIdx.x;
    if (t >= total) return;
    const int lane = t & 31;
    const int ks   = (t >> 5) % nkb;
    const int tm   = (t >> 5) / nkb;
    const int hi   = lane >> 4, lo = lane & 15;
    const int row  = tm * 16 + lo;
    const float* base = X + (size_t)row * Cpad + ks * 32 + hi * 8;
    const float4 a0 = *(const float4*)(base);
    const float4 a1 = *(const float4*)(base + 4);
    const float4 a2 = *(const float4*)(base + 16);
    const float4 a3 = *(const float4*)(base + 20);
    float f[16] = {a0.x, a0.y, a0.z, a0.w, a1.x, a1.y, a1.z, a1.w,
                   a2.x, a2.y, a2.z, a2.w, a3.x, a3.y, a3.z, a3.w};
    if (ACT != ACT_NONE) {
        const float* sb = scale + ks * 32 + hi * 8;
        const float* hb = shift + ks * 32 + hi * 8;
        const float4 s0 = *(const float4*)(sb),      s1 = *(const float4*)(sb + 4);
        const float4 s2 = *(const float4*)(sb + 16), s3 = *(const float4*)(sb + 20);
        const float4 h0 = *(const float4*)(hb),      h1 = *(const float4*)(hb + 4);
        const float4 h2 = *(const float4*)(hb + 16), h3 = *(const float4*)(hb + 20);
        const float sc[16] = {s0.x, s0.y, s0.z, s0.w, s1.x, s1.y, s1.z, s1.w,
                              s2.x, s2.y, s2.z, s2.w, s3.x, s3.y, s3.z, s3.w};
        const float sh[16] = {h0.x, h0.y, h0.z, h0.w, h1.x, h1.y, h1.z, h1.w,
                              h2.x, h2.y, h2.z, h2.w, h3.x, h3.y, h3.z, h3.w};
        #pragma unroll
        for (int e = 0; e < 16; ++e) {
            float v = fmaf(f[e], sc[e], sh[e]);
            f[e] = (ACT == ACT_RELU) ? fmaxf(v, 0.0f) : ((v > 0.0f) ? v : 0.2f * v);
        }
    }
    v16h o;
    #pragma unroll
    for (int e = 0; e < 16; ++e) o[e] = (_Float16)f[e];
    ((v16h*)Apk)[t] = o;
}

// ------------------------ weight (B) fragment pack -------------------------
__global__ __launch_bounds__(256)
void bpack_kernel(const float* __restrict__ W, int Cout, int Cin, int nkb,
                  _Float16* __restrict__ Bpk, int total) {
    const int t = blockIdx.x * blockDim.x + threadIdx.x;
    if (t >= total) return;
    const int lane = t & 31;
    const int ks   = (t >> 5) % nkb;
    const int hi   = lane >> 4, lo = lane & 15;
    const int tn   = (t >> 5) / nkb;
    const int n    = tn * 16 + lo;
    const int nc   = n < Cout ? n : Cout - 1;
    v16h o;
    #pragma unroll
    for (int e = 0; e < 16; ++e) {
        const int kk = ks * 32 + (e & 7) + 8 * hi + 16 * (e >> 3);
        const int kc = kk < Cin ? kk : Cin - 1;
        float v = W[(size_t)nc * Cin + kc];
        if (n >= Cout || kk >= Cin) v = 0.0f;
        o[e] = (_Float16)v;
    }
    ((v16h*)Bpk)[t] = o;
}

__global__ void pad_bias(const float* __restrict__ b, int C,
                         float* __restrict__ out, int Cp) {
    const int t = blockIdx.x * blockDim.x + threadIdx.x;
    if (t < Cp) out[t] = (t < C) ? b[t] : 0.0f;
}

// ------------------------------ WMMA GEMM ----------------------------------
// Guard-free: M % 16 == 0, padded operands, padded bias, ldo % 16 == 0.
template <bool MAXPOOL16>
__global__ __launch_bounds__(128)
void wmma_gemm_packed(const _Float16* __restrict__ Apk, const _Float16* __restrict__ Bpk,
                      const float* __restrict__ biasp, int nkb,
                      int tm_count, int tn_count, int ldo, float* __restrict__ OUT) {
    const int wave = threadIdx.x >> 5;
    const int lane = threadIdx.x & 31;
    const int tile = blockIdx.x * 4 + wave;            // wave-uniform
    if (tile >= tm_count * tn_count) return;           // whole-wave exit
    const int tm = tile / tn_count;
    const int tn = tile % tn_count;
    const int hi = lane >> 4, lo = lane & 15;

    const v16h* A = (const v16h*)Apk + ((size_t)tm * nkb * 32 + lane);
    const v16h* Bv = (const v16h*)Bpk + ((size_t)tn * nkb * 32 + lane);
    v8f acc = {};
    for (int ks = 0; ks < nkb; ++ks) {
        const v16h a = A[(size_t)ks * 32];
        const v16h b = Bv[(size_t)ks * 32];
        acc = __builtin_amdgcn_wmma_f32_16x16x32_f16(
            /*neg_a=*/false, a, /*neg_b=*/false, b,
            /*c_mod=*/(short)0, acc, /*reuse_a=*/false, /*reuse_b=*/false);
    }
    const float bv = biasp[tn * 16 + lo];
    if (MAXPOOL16) {   // one M-tile == one K=16 neighbor group
        float mx = acc[0] + bv;
        #pragma unroll
        for (int r = 1; r < 8; ++r) mx = fmaxf(mx, acc[r] + bv);
        mx = fmaxf(mx, __shfl_xor(mx, 16, 32));
        if (hi == 0) OUT[(size_t)tm * ldo + tn * 16 + lo] = mx;
    } else {
        #pragma unroll
        for (int r = 0; r < 8; ++r)
            OUT[((size_t)tm * 16 + r + 8 * hi) * ldo + tn * 16 + lo] = acc[r] + bv;
    }
}

// ------------------------- row max-pool (SA3 global) -----------------------
__global__ void maxpool_rows(const float* __restrict__ H, int B_, int rows, int C,
                             float* __restrict__ out) {
    const int t = blockIdx.x * blockDim.x + threadIdx.x;
    if (t >= B_ * C) return;
    const int b = t / C, c = t % C;
    float m = -3.0e38f;
    for (int r = 0; r < rows; ++r)
        m = fmaxf(m, H[((size_t)b * rows + r) * C + c]);
    out[t] = m;
}

__global__ void copy_col0(const float* __restrict__ F, int ld, int M,
                          float* __restrict__ out) {
    const int t = blockIdx.x * blockDim.x + threadIdx.x;
    if (t < M) out[t] = F[(size_t)t * ld];
}

// ------------------------------ host side ----------------------------------
struct Bump {
    uintptr_t p;
    void* take(size_t bytes) {
        void* r = (void*)p;
        p += (bytes + 255) & ~(size_t)255;
        return r;
    }
};

static inline int cdiv(int a, int b) { return (a + b - 1) / b; }

template <bool MP>
static void run_gemm(const _Float16* A, const _Float16* Bp, const float* biasp,
                     int nkb, int tm, int tn, int ldo, float* OUT, hipStream_t s) {
    const int tiles = tm * tn;
    wmma_gemm_packed<MP><<<cdiv(tiles, 4), 128, 0, s>>>(A, Bp, biasp, nkb, tm, tn, ldo, OUT);
}

static void run_bpack(const float* W, int Cout, int Cin, int CoutPad, int CinPad,
                      _Float16* Bpk, hipStream_t s) {
    const int nkb = CinPad / 32;
    const int total = (CoutPad / 16) * nkb * 32;
    bpack_kernel<<<cdiv(total, 256), 256, 0, s>>>(W, Cout, Cin, nkb, Bpk, total);
}

extern "C" void kernel_launch(void* const* d_in, const int* in_sizes, int n_in,
                              void* d_out, int out_size, void* d_ws, size_t ws_size,
                              hipStream_t stream) {
    (void)in_sizes; (void)n_in; (void)out_size; (void)ws_size;
    constexpr int Bb = 16, Np = 8192, S1 = 512, S2 = 128;

    const float* points    = (const float*)d_in[0];
    const float* sa1_c1_w  = (const float*)d_in[1];
    const float* sa1_c1_b  = (const float*)d_in[2];
    const float* sa1_bn_g  = (const float*)d_in[3];
    const float* sa1_bn_be = (const float*)d_in[4];
    const float* sa1_c2_w  = (const float*)d_in[5];
    const float* sa1_c2_b  = (const float*)d_in[6];
    const float* sa2_c1_w  = (const float*)d_in[7];
    const float* sa2_c1_b  = (const float*)d_in[8];
    const float* sa2_bn_g  = (const float*)d_in[9];
    const float* sa2_bn_be = (const float*)d_in[10];
    const float* sa2_c2_w  = (const float*)d_in[11];
    const float* sa2_c2_b  = (const float*)d_in[12];
    const float* sa3_c1_w  = (const float*)d_in[13];
    const float* sa3_c1_b  = (const float*)d_in[14];
    const float* sa3_bn_g  = (const float*)d_in[15];
    const float* sa3_bn_be = (const float*)d_in[16];
    const float* sa3_c2_w  = (const float*)d_in[17];
    const float* sa3_c2_b  = (const float*)d_in[18];
    const float* fc1_w     = (const float*)d_in[19];
    const float* fc1_b     = (const float*)d_in[20];
    const float* fc1_bn_g  = (const float*)d_in[21];
    const float* fc1_bn_be = (const float*)d_in[22];
    const float* fc2_w     = (const float*)d_in[23];
    const float* fc2_b     = (const float*)d_in[24];
    const float* fc2_bn_g  = (const float*)d_in[25];
    const float* fc2_bn_be = (const float*)d_in[26];
    const float* fc3_w     = (const float*)d_in[27];
    const float* fc3_b     = (const float*)d_in[28];

    Bump ws{(uintptr_t)d_ws};
    // -------- persistent buffers --------
    int*   fidx1  = (int*)ws.take((size_t)Bb * S1 * 4);
    float* l1_xyz = (float*)ws.take((size_t)Bb * S1 * 3 * 4);
    int*   knn1   = (int*)ws.take((size_t)Bb * S1 * 16 * 4);
    float* l1_p   = (float*)ws.take((size_t)Bb * S1 * 128 * 4);
    int*   fidx2  = (int*)ws.take((size_t)Bb * S2 * 4);
    float* l2_xyz = (float*)ws.take((size_t)Bb * S2 * 3 * 4);
    int*   knn2   = (int*)ws.take((size_t)Bb * S2 * 16 * 4);
    float* l2_p   = (float*)ws.take((size_t)Bb * S2 * 256 * 4);
    float* gfeat  = (float*)ws.take((size_t)Bb * 256 * 4);
    float* F1     = (float*)ws.take((size_t)Bb * 256 * 4);
    float* F2     = (float*)ws.take((size_t)Bb * 128 * 4);
    float* F3     = (float*)ws.take((size_t)Bb * 16 * 4);
    float* scl1   = (float*)ws.take(64 * 4);
    float* shf1   = (float*)ws.take(64 * 4);
    float* scl2   = (float*)ws.take(128 * 4);
    float* shf2   = (float*)ws.take(128 * 4);
    float* scl3   = (float*)ws.take(256 * 4);
    float* shf3   = (float*)ws.take(256 * 4);
    float* sclf1  = (float*)ws.take(256 * 4);
    float* shff1  = (float*)ws.take(256 * 4);
    float* sclf2  = (float*)ws.take(128 * 4);
    float* shff2  = (float*)ws.take(128 * 4);
    const uintptr_t transient = ws.p;

    // ================= SA1: FPS 8192->512, KNN16, MLP 6->64->128 ==========
    {
        ws.p = transient;
        const int nq = Bb * S1;           // 8192 groups
        const int M  = nq * 16;           // 131072 rows, tm = 8192
        _Float16* Apk1 = (_Float16*)ws.take((size_t)M * 32 * 2);      // Cpad 32
        _Float16* Bpk1 = (_Float16*)ws.take((size_t)4 * 1 * 32 * 16 * 2);
        float*    bp1  = (float*)ws.take(64 * 4);
        float*    H1   = (float*)ws.take((size_t)M * 64 * 4);
        _Float16* Apk2 = (_Float16*)ws.take((size_t)M * 64 * 2);
        _Float16* Bpk2 = (_Float16*)ws.take((size_t)8 * 2 * 32 * 16 * 2);
        float*    bp2  = (float*)ws.take(128 * 4);

        fps_kernel<<<Bb, FPS_THREADS, 0, stream>>>(points, Np, S1, fidx1, l1_xyz);
        knn_kernel<<<cdiv(nq, 8), 256, 0, stream>>>(l1_xyz, points, S1, Np, nq, knn1);
        build_sa1_packed<<<cdiv(M, 256), 256, 0, stream>>>(points, l1_xyz, knn1, Apk1, nq, Np);
        run_bpack(sa1_c1_w, 64, 6, 64, 32, Bpk1, stream);
        pad_bias<<<1, 64, 0, stream>>>(sa1_c1_b, 64, bp1, 64);
        run_gemm<false>(Apk1, Bpk1, bp1, 1, M / 16, 4, 64, H1, stream);
        stats_kernel<<<64, 256, 0, stream>>>(H1, M, 64, sa1_bn_g, sa1_bn_be, scl1, shf1);
        {
            const int total = (M / 16) * 2 * 32;
            apack_kernel<ACT_RELU><<<cdiv(total, 256), 256, 0, stream>>>(
                H1, 2, 64, scl1, shf1, Apk2, total);
        }
        run_bpack(sa1_c2_w, 128, 64, 128, 64, Bpk2, stream);
        pad_bias<<<1, 128, 0, stream>>>(sa1_c2_b, 128, bp2, 128);
        run_gemm<true>(Apk2, Bpk2, bp2, 2, M / 16, 8, 128, l1_p, stream);
    }

    // ================= SA2: FPS 512->128, KNN16, MLP 131->128->256 ========
    {
        ws.p = transient;
        const int nq2 = Bb * S2;          // 2048 groups
        const int M   = nq2 * 16;         // 32768 rows
        const int nkb1 = 5;               // Cpad 160
        _Float16* Apk1 = (_Float16*)ws.take((size_t)M * 160 * 2);
        _Float16* Bpk1 = (_Float16*)ws.take((size_t)8 * nkb1 * 32 * 16 * 2);
        float*    bp1  = (float*)ws.take(128 * 4);
        float*    H2   = (float*)ws.take((size_t)M * 128 * 4);
        _Float16* Apk2 = (_Float16*)ws.take((size_t)M * 128 * 2);
        _Float16* Bpk2 = (_Float16*)ws.take((size_t)16 * 4 * 32 * 16 * 2);
        float*    bp2  = (float*)ws.take(256 * 4);

        fps_kernel<<<Bb, FPS_THREADS, 0, stream>>>(l1_xyz, S1, S2, fidx2, l2_xyz);
        knn_kernel<<<cdiv(nq2, 8), 256, 0, stream>>>(l2_xyz, l1_xyz, S2, S1, nq2, knn2);
        build_sa2_packed<<<cdiv(M * nkb1, 256), 256, 0, stream>>>(
            l1_xyz, l1_p, l2_xyz, knn2, Apk1, M, nkb1, S1, S2, 128);
        run_bpack(sa2_c1_w, 128, 131, 128, 160, Bpk1, stream);
        pad_bias<<<1, 128, 0, stream>>>(sa2_c1_b, 128, bp1, 128);
        run_gemm<false>(Apk1, Bpk1, bp1, nkb1, M / 16, 8, 128, H2, stream);
        stats_kernel<<<128, 256, 0, stream>>>(H2, M, 128, sa2_bn_g, sa2_bn_be, scl2, shf2);
        {
            const int total = (M / 16) * 4 * 32;
            apack_kernel<ACT_RELU><<<cdiv(total, 256), 256, 0, stream>>>(
                H2, 4, 128, scl2, shf2, Apk2, total);
        }
        run_bpack(sa2_c2_w, 256, 128, 256, 128, Bpk2, stream);
        pad_bias<<<1, 256, 0, stream>>>(sa2_c2_b, 256, bp2, 256);
        run_gemm<true>(Apk2, Bpk2, bp2, 4, M / 16, 16, 256, l2_p, stream);
    }

    // ================= SA3: group-all, MLP 259->256->256, max 128 =========
    {
        ws.p = transient;
        const int M = Bb * S2;            // 2048 rows
        const int nkb1 = 9;               // Cpad 288
        _Float16* Apk1 = (_Float16*)ws.take((size_t)M * 288 * 2);
        _Float16* Bpk1 = (_Float16*)ws.take((size_t)16 * nkb1 * 32 * 16 * 2);
        float*    bp1  = (float*)ws.take(256 * 4);
        float*    H3   = (float*)ws.take((size_t)M * 256 * 4);
        _Float16* Apk2 = (_Float16*)ws.take((size_t)M * 256 * 2);
        _Float16* Bpk2 = (_Float16*)ws.take((size_t)16 * 8 * 32 * 16 * 2);
        float*    bp2  = (float*)ws.take(256 * 4);
        float*    H3b  = (float*)ws.take((size_t)M * 256 * 4);

        build_sa3_packed<<<cdiv(M * nkb1, 256), 256, 0, stream>>>(
            l2_xyz, l2_p, Apk1, M, nkb1, 256);
        run_bpack(sa3_c1_w, 256, 259, 256, 288, Bpk1, stream);
        pad_bias<<<1, 256, 0, stream>>>(sa3_c1_b, 256, bp1, 256);
        run_gemm<false>(Apk1, Bpk1, bp1, nkb1, M / 16, 16, 256, H3, stream);
        stats_kernel<<<256, 256, 0, stream>>>(H3, M, 256, sa3_bn_g, sa3_bn_be, scl3, shf3);
        {
            const int total = (M / 16) * 8 * 32;
            apack_kernel<ACT_RELU><<<cdiv(total, 256), 256, 0, stream>>>(
                H3, 8, 256, scl3, shf3, Apk2, total);
        }
        run_bpack(sa3_c2_w, 256, 256, 256, 256, Bpk2, stream);
        pad_bias<<<1, 256, 0, stream>>>(sa3_c2_b, 256, bp2, 256);
        run_gemm<false>(Apk2, Bpk2, bp2, 8, M / 16, 16, 256, H3b, stream);
        maxpool_rows<<<cdiv(Bb * 256, 256), 256, 0, stream>>>(H3b, Bb, S2, 256, gfeat);
    }

    // ================= FC head: 256 -> 256 -> 128 -> 1 ====================
    {
        ws.p = transient;
        _Float16* ApkA = (_Float16*)ws.take((size_t)16 * 256 * 2);
        _Float16* BpkA = (_Float16*)ws.take((size_t)16 * 8 * 32 * 16 * 2);
        float*    bpA  = (float*)ws.take(256 * 4);
        _Float16* ApkB = (_Float16*)ws.take((size_t)16 * 256 * 2);
        _Float16* BpkB = (_Float16*)ws.take((size_t)8 * 8 * 32 * 16 * 2);
        float*    bpB  = (float*)ws.take(128 * 4);
        _Float16* ApkC = (_Float16*)ws.take((size_t)16 * 128 * 2);
        _Float16* BpkC = (_Float16*)ws.take((size_t)1 * 4 * 32 * 16 * 2);
        float*    bpC  = (float*)ws.take(16 * 4);

        // fc1
        {
            const int total = 1 * 8 * 32;
            apack_kernel<ACT_NONE><<<1, 256, 0, stream>>>(gfeat, 8, 256,
                                                          nullptr, nullptr, ApkA, total);
        }
        run_bpack(fc1_w, 256, 256, 256, 256, BpkA, stream);
        pad_bias<<<1, 256, 0, stream>>>(fc1_b, 256, bpA, 256);
        run_gemm<false>(ApkA, BpkA, bpA, 8, 1, 16, 256, F1, stream);
        stats_kernel<<<256, 256, 0, stream>>>(F1, 16, 256, fc1_bn_g, fc1_bn_be, sclf1, shff1);
        // fc2
        {
            const int total = 1 * 8 * 32;
            apack_kernel<ACT_LEAKY><<<1, 256, 0, stream>>>(F1, 8, 256,
                                                           sclf1, shff1, ApkB, total);
        }
        run_bpack(fc2_w, 128, 256, 128, 256, BpkB, stream);
        pad_bias<<<1, 128, 0, stream>>>(fc2_b, 128, bpB, 128);
        run_gemm<false>(ApkB, BpkB, bpB, 8, 1, 8, 128, F2, stream);
        stats_kernel<<<128, 256, 0, stream>>>(F2, 16, 128, fc2_bn_g, fc2_bn_be, sclf2, shff2);
        // fc3 (Cout 1 -> padded 16, copy column 0)
        {
            const int total = 1 * 4 * 32;
            apack_kernel<ACT_LEAKY><<<1, 128, 0, stream>>>(F2, 4, 128,
                                                           sclf2, shff2, ApkC, total);
        }
        run_bpack(fc3_w, 1, 128, 16, 128, BpkC, stream);
        pad_bias<<<1, 16, 0, stream>>>(fc3_b, 1, bpC, 16);
        run_gemm<false>(ApkC, BpkC, bpC, 4, 1, 1, 16, F3, stream);
        copy_col0<<<1, 16, 0, stream>>>(F3, 16, 16, (float*)d_out);
    }
}
